// Attention_4380866642483
// MI455X (gfx1250) — matrix operand order repaired and checked by reading, compiled, not run on hardware
//
#include <hip/hip_runtime.h>

#define BB 64
#define TT 2048
#define ENCD 512
#define DECD 1024
#define CONVC 32
#define KK 31
#define HIDD 256
#define PADW 15

typedef __attribute__((ext_vector_type(16))) __bf16 v16bf;
typedef __attribute__((ext_vector_type(8)))  float  v8f;

union BF16Bits { __bf16 h; unsigned short u; };

// fp32 -> bf16 round-to-nearest-even via integer ops
__device__ __forceinline__ __bf16 f2bf(float x) {
  union { float f; unsigned u; } a; a.f = x;
  unsigned r = a.u + 0x7fffu + ((a.u >> 16) & 1u);
  BF16Bits b; b.u = (unsigned short)(r >> 16);
  return b.h;
}

// ---------------- K0a: fp32 -> bf16 conversion (W_enc, W_att) ----------------
__global__ void cvt_bf16_kernel(const float* __restrict__ w,
                                unsigned short* __restrict__ o, int n) {
  int i = blockIdx.x * blockDim.x + threadIdx.x;
  if (i < n) {
    BF16Bits b; b.h = f2bf(w[i]);
    o[i] = b.u;
  }
}

// ---------------- K0b: dec_proj[b][h] = dec_state[b] . W_dec[h] ----------------
__global__ __launch_bounds__(HIDD) void dec_proj_kernel(
    const float* __restrict__ dec_state, const float* __restrict__ W_dec,
    float* __restrict__ out) {
  int b = blockIdx.x, h = threadIdx.x;
  const float* ds = dec_state + (size_t)b * DECD;
  const float* wr = W_dec + (size_t)h * DECD;
  float acc = 0.f;
  for (int d = 0; d < DECD; d += 4) {
    float4 a = *(const float4*)(ds + d);
    float4 w = *(const float4*)(wr + d);
    acc += a.x * w.x + a.y * w.y + a.z * w.z + a.w * w.w;
  }
  out[b * HIDD + h] = acc;
}

// ---------------- K1: fused energy kernel -------------------------------------
// grid = (TT/16, BB), block = 128 (4 waves). Wave w owns n-tiles [4w .. 4w+3]
// (h range [64w, 64w+64)); tile is 16 t-rows x 256 h.
// pm GEMM: K=ENC=512 via 16 bf16 WMMA steps; att_conv GEMM: K=CONV=32 via
// ONE bf16 WMMA per n-tile, accumulated into the same C/D registers.
__global__ __launch_bounds__(128) void energy_kernel(
    const float* __restrict__ enc, const float* __restrict__ prev,
    const int* __restrict__ text_len, const unsigned short* __restrict__ wencbf,
    const unsigned short* __restrict__ wattbf,
    const float* __restrict__ decP, const float* __restrict__ benc,
    const float* __restrict__ convw,
    const float* __restrict__ ww, const float* __restrict__ wb,
    float* __restrict__ energy) {
  const int t0   = blockIdx.x * 16;
  const int b    = blockIdx.y;
  const int tid  = threadIdx.x;
  const int lane = tid & 31;
  const int wv   = tid >> 5;
  const int nl   = lane & 15;
  const int half = lane >> 4;

  __shared__ float decS[HIDD];
  __shared__ float wwS[HIDD];
  __shared__ float convS[CONVC * 16];     // convS[c*16 + m]
  __shared__ float redS[4][16];

  // Stage dec-term (+ b_enc) and w_w
  for (int i = tid; i < HIDD; i += 128) {
    decS[i] = decP[b * HIDD + i] + benc[i];
    wwS[i]  = ww[i];
  }
  // Location conv: convS[c][m] = sum_k prev[b, t0+m-PAD+k] * conv_w[c,0,k]
  for (int i = tid; i < CONVC * 16; i += 128) {
    int c = i % CONVC, m = i / CONVC;
    int tg = t0 + m;
    float s = 0.f;
    #pragma unroll
    for (int k = 0; k < KK; ++k) {
      int p = tg - PADW + k;
      if (p >= 0 && p < TT) s += prev[(size_t)b * TT + p] * convw[c * KK + k];
    }
    convS[c * 16 + m] = s;
  }
  __syncthreads();

  // ---- main GEMM: pm tile via bf16 WMMA (K = 512) ----
  v8f acc[4] = {};
  const int n0 = wv * 64;
  const float* arow = enc + ((size_t)(b * TT + t0 + nl)) * ENCD;

  for (int kb = 0; kb < ENCD; kb += 32) {
    // A fragment (16x32 bf16): lane (m=nl, half) holds K runs [8h..8h+7], [16+8h..23+8h]
    v16bf av;
    const float* ar = arow + kb + 8 * half;
    #pragma unroll
    for (int j = 0; j < 8; ++j) {
      av[j]     = f2bf(ar[j]);
      av[8 + j] = f2bf(ar[16 + j]);
    }
    #pragma unroll
    for (int j = 0; j < 4; ++j) {
      // B fragment (32x16 bf16): lane (n=nl, half) holds K runs [8h..8h+7], [16+8h..23+8h]
      union { v16bf v; uint4 q[2]; } bf;
      const unsigned short* bp =
          wencbf + (size_t)(n0 + 16 * j + nl) * ENCD + kb + 8 * half;
      bf.q[0] = *(const uint4*)(bp);
      bf.q[1] = *(const uint4*)(bp + 16);
      acc[j] = __builtin_amdgcn_wmma_f32_16x16x32_bf16(
          false, av, false, bf.v, (short)0, acc[j], false, false);
    }
  }

  // ---- att_conv GEMM: one WMMA per n-tile (K = CONV = 32), same accumulators ----
  v16bf cav;
  #pragma unroll
  for (int j = 0; j < 8; ++j) {
    cav[j]     = f2bf(convS[(8 * half + j) * 16 + nl]);        // k = 8h + j
    cav[8 + j] = f2bf(convS[(16 + 8 * half + j) * 16 + nl]);   // k = 16 + 8h + j
  }
  #pragma unroll
  for (int j = 0; j < 4; ++j) {
    union { v16bf v; uint4 q[2]; } bf;
    const unsigned short* bp =
        wattbf + (size_t)(n0 + 16 * j + nl) * CONVC + 8 * half;
    bf.q[0] = *(const uint4*)(bp);
    bf.q[1] = *(const uint4*)(bp + 16);
    acc[j] = __builtin_amdgcn_wmma_f32_16x16x32_bf16(
        false, cav, false, bf.v, (short)0, acc[j], false, false);
  }

  // ---- epilogue: + dec, tanh, dot with w_w, reduce over h ----
  float part[8] = {0.f, 0.f, 0.f, 0.f, 0.f, 0.f, 0.f, 0.f};
  #pragma unroll
  for (int j = 0; j < 4; ++j) {
    const int h = n0 + 16 * j + nl;
    const float dech = decS[h];
    const float wwh  = wwS[h];
    #pragma unroll
    for (int r = 0; r < 8; ++r)
      part[r] += wwh * tanhf(acc[j][r] + dech);
  }

  // cross-lane reduce within each 16-lane half (masks 1,2,4,8 keep the half bit)
  #pragma unroll
  for (int r = 0; r < 8; ++r) {
    float v = part[r];
    v += __shfl_xor(v, 1, 32);
    v += __shfl_xor(v, 2, 32);
    v += __shfl_xor(v, 4, 32);
    v += __shfl_xor(v, 8, 32);
    part[r] = v;
  }
  if (nl == 0) {
    #pragma unroll
    for (int r = 0; r < 8; ++r) redS[wv][half * 8 + r] = part[r];
  }
  __syncthreads();

  if (tid < 16) {
    const int m = tid;
    float e = redS[0][m] + redS[1][m] + redS[2][m] + redS[3][m] + wb[0];
    const int t = t0 + m;
    if (t >= text_len[b]) e = -__builtin_inff();
    energy[(size_t)b * TT + t] = e;
  }
}

// ---------------- K2: masked softmax over T per batch row ----------------
__global__ __launch_bounds__(256) void softmax_kernel(
    const float* __restrict__ energy, float* __restrict__ attw) {
  int b = blockIdx.x;
  __shared__ float red[256];
  const float* e = energy + (size_t)b * TT;
  float mx = -__builtin_inff();
  for (int t = threadIdx.x; t < TT; t += 256) mx = fmaxf(mx, e[t]);
  red[threadIdx.x] = mx; __syncthreads();
  for (int s = 128; s > 0; s >>= 1) {
    if (threadIdx.x < s) red[threadIdx.x] = fmaxf(red[threadIdx.x], red[threadIdx.x + s]);
    __syncthreads();
  }
  mx = red[0]; __syncthreads();
  float sm = 0.f;
  for (int t = threadIdx.x; t < TT; t += 256) sm += expf(e[t] - mx);  // -inf -> 0
  red[threadIdx.x] = sm; __syncthreads();
  for (int s = 128; s > 0; s >>= 1) {
    if (threadIdx.x < s) red[threadIdx.x] += red[threadIdx.x + s];
    __syncthreads();
  }
  const float inv = 1.f / red[0];
  for (int t = threadIdx.x; t < TT; t += 256)
    attw[(size_t)b * TT + t] = expf(e[t] - mx) * inv;
}

// ---------------- K3: att_c[b][c] = sum_t enc[b][t][c] * attw[b][t] ----------------
__global__ __launch_bounds__(256) void context_kernel(
    const float* __restrict__ enc, const float* __restrict__ attw,
    float* __restrict__ attc) {
  const int b = blockIdx.x;
  const int c = blockIdx.y * 256 + threadIdx.x;
  __shared__ float wS[128];
  float acc = 0.f;
  for (int t0 = 0; t0 < TT; t0 += 128) {
    __syncthreads();
    if (threadIdx.x < 128) wS[threadIdx.x] = attw[(size_t)b * TT + t0 + threadIdx.x];
    __syncthreads();
    const float* base = enc + ((size_t)(b * TT + t0)) * ENCD + c;
    #pragma unroll 8
    for (int t = 0; t < 128; ++t) acc += base[(size_t)t * ENCD] * wS[t];
  }
  attc[b * ENCD + c] = acc;
}

extern "C" void kernel_launch(void* const* d_in, const int* in_sizes, int n_in,
                              void* d_out, int out_size, void* d_ws, size_t ws_size,
                              hipStream_t stream) {
  const float* enc       = (const float*)d_in[0];
  const float* dec_state = (const float*)d_in[1];
  const float* prev      = (const float*)d_in[2];
  const int*   text_len  = (const int*)d_in[3];
  const float* W_enc     = (const float*)d_in[4];
  const float* b_enc     = (const float*)d_in[5];
  const float* W_dec     = (const float*)d_in[6];
  const float* W_att     = (const float*)d_in[7];
  const float* conv_w    = (const float*)d_in[8];
  const float* w_w       = (const float*)d_in[9];
  const float* w_b       = (const float*)d_in[10];

  float* out   = (float*)d_out;
  float* att_c = out;               // (B, ENC) first in return order
  float* att_w = out + BB * ENCD;   // (B, T)

  char* ws = (char*)d_ws;
  unsigned short* wencbf = (unsigned short*)ws;                  // 262144 B
  unsigned short* wattbf = (unsigned short*)(ws + 262144);       //  16384 B
  float* decP   = (float*)(ws + 262144 + 16384);                 //  65536 B
  float* energy = (float*)(ws + 262144 + 16384 + 65536);         // 524288 B

  cvt_bf16_kernel<<<(HIDD * ENCD + 255) / 256, 256, 0, stream>>>(W_enc, wencbf, HIDD * ENCD);
  cvt_bf16_kernel<<<(HIDD * CONVC + 255) / 256, 256, 0, stream>>>(W_att, wattbf, HIDD * CONVC);
  dec_proj_kernel<<<BB, HIDD, 0, stream>>>(dec_state, W_dec, decP);
  energy_kernel<<<dim3(TT / 16, BB), 128, 0, stream>>>(
      enc, prev, text_len, wencbf, wattbf, decP, b_enc, conv_w, w_w, w_b, energy);
  softmax_kernel<<<BB, 256, 0, stream>>>(energy, att_w);
  context_kernel<<<dim3(BB, ENCD / 256), 256, 0, stream>>>(enc, att_w, att_c);
}
